// PFM_18279380812280
// MI455X (gfx1250) — matrix operand order repaired and checked by reading, not compiled, not run on hardware
//
#include <hip/hip_runtime.h>

// ---- problem constants ------------------------------------------------------
#define BQ 8
#define LQ 1024
#define HQ 1024
#define MQ 128
#define VQ 32
#define V2 1024            // VQ*VQ
#define LN_EPS 1e-5f

// ---- CDNA5 WMMA types -------------------------------------------------------
typedef __attribute__((ext_vector_type(16))) __bf16          v16bf;
typedef __attribute__((ext_vector_type(8)))  float           v8f;
typedef __attribute__((ext_vector_type(8)))  unsigned short  us8;
typedef __attribute__((ext_vector_type(16))) unsigned short  us16;

__device__ __forceinline__ unsigned short f2bf(float f) {
  union { float f; unsigned u; } x; x.f = f;
  unsigned r = x.u + 0x7fffu + ((x.u >> 16) & 1u);   // round-to-nearest-even
  return (unsigned short)(r >> 16);
}
__device__ __forceinline__ float bf2f(unsigned short b) {
  union { unsigned u; float f; } x; x.u = ((unsigned)b) << 16;
  return x.f;
}

// Build a 16-lane bf16 WMMA fragment from two 16-byte LDS/global reads.
__device__ __forceinline__ v16bf load_frag(const unsigned short* p0,
                                           const unsigned short* p1) {
  union { v16bf b; struct { us8 lo; us8 hi; } s; } f;
  f.s.lo = *(const us8*)p0;
  f.s.hi = *(const us8*)p1;
  return f.b;
}

__device__ __forceinline__ v8f wmma_bf16(v16bf a, v16bf b, v8f c) {
  return __builtin_amdgcn_wmma_f32_16x16x32_bf16(false, a, false, b,
                                                 (short)0, c, false, false);
}

// ---- 1) LayerNorm (f32 in -> bf16 out) -------------------------------------
__global__ void __launch_bounds__(256)
ln_kernel(const float* __restrict__ x, const float* __restrict__ gamma,
          const float* __restrict__ beta, unsigned short* __restrict__ xo) {
  __shared__ float s1[256], s2[256];
  const int row = blockIdx.x, t = threadIdx.x;
  const float* xr = x + (size_t)row * HQ;
  float4 v = *(const float4*)(xr + t * 4);
  s1[t] = v.x + v.y + v.z + v.w;
  s2[t] = v.x*v.x + v.y*v.y + v.z*v.z + v.w*v.w;
  __syncthreads();
  for (int off = 128; off > 0; off >>= 1) {
    if (t < off) { s1[t] += s1[t + off]; s2[t] += s2[t + off]; }
    __syncthreads();
  }
  const float mu  = s1[0] * (1.0f / HQ);
  const float var = s2[0] * (1.0f / HQ) - mu * mu;
  const float rs  = rsqrtf(var + LN_EPS);
  float4 g = *(const float4*)(gamma + t * 4);
  float4 bt = *(const float4*)(beta + t * 4);
  unsigned short* o = xo + (size_t)row * HQ + t * 4;
  o[0] = f2bf((v.x - mu) * rs * g.x + bt.x);
  o[1] = f2bf((v.y - mu) * rs * g.y + bt.y);
  o[2] = f2bf((v.z - mu) * rs * g.z + bt.z);
  o[3] = f2bf((v.w - mu) * rs * g.w + bt.w);
}

// ---- 2) f32 -> bf16 weight conversion --------------------------------------
__global__ void __launch_bounds__(256)
f32_to_bf16(const float* __restrict__ in, unsigned short* __restrict__ out, int n) {
  for (int i = blockIdx.x * blockDim.x + threadIdx.x; i < n;
       i += gridDim.x * blockDim.x)
    out[i] = f2bf(in[i]);
}

// ---- 3) q/k GEMM: gathered x_ln rows @ Wq^T / Wk^T, bf16 out ---------------
__global__ void __launch_bounds__(256)
qk_gemm(const unsigned short* __restrict__ xln,
        const unsigned short* __restrict__ wq,
        const unsigned short* __restrict__ wk,
        const int* __restrict__ midx,
        unsigned short* __restrict__ qout,
        unsigned short* __restrict__ kout) {
  __shared__ unsigned short As[128][72];   // padded: 72*2B = 144B rows (16B-aligned, conflict-free)
  __shared__ unsigned short Ws[128][72];
  const int t = threadIdx.x;
  const int rowtile = blockIdx.x, ntile = blockIdx.y, which = blockIdx.z;
  const unsigned short* W = which ? wk : wq;
  unsigned short* dst = which ? kout : qout;
  const int wid = t >> 5, lane = t & 31, half = lane >> 4, hm = lane & 15;
  const int wm = wid & 3, wn = wid >> 2;

  v8f acc[2][4];
  const v8f vz = {0.f,0.f,0.f,0.f,0.f,0.f,0.f,0.f};
  for (int a = 0; a < 2; ++a) for (int b = 0; b < 4; ++b) acc[a][b] = vz;

  for (int k0 = 0; k0 < HQ; k0 += 64) {
    for (int pass = 0; pass < 4; ++pass) {
      const int idx = pass * 2048 + t * 8;
      const int j = idx >> 6, col = idx & 63;
      const int R = rowtile * 128 + j;
      const int bb = R >> 7, mm = R & 127;
      const int src = midx[bb * MQ + mm];
      *(us8*)&As[j][col] = *(const us8*)(xln + ((size_t)bb * LQ + src) * HQ + k0 + col);
      *(us8*)&Ws[j][col] = *(const us8*)(W + (size_t)(ntile * 128 + j) * HQ + k0 + col);
    }
    __syncthreads();
    for (int kk = 0; kk < 64; kk += 32) {
      const int r0 = wm * 32 + hm, r1 = r0 + 16;
      v16bf a0 = load_frag(&As[r0][kk + half*8], &As[r0][kk + 16 + half*8]);
      v16bf a1 = load_frag(&As[r1][kk + half*8], &As[r1][kk + 16 + half*8]);
#pragma unroll
      for (int nt = 0; nt < 4; ++nt) {
        const unsigned short* bp = &Ws[wn*64 + nt*16 + hm][kk + half*8];
        v16bf bf = load_frag(bp, bp + 16);
        acc[0][nt] = wmma_bf16(a0, bf, acc[0][nt]);
        acc[1][nt] = wmma_bf16(a1, bf, acc[1][nt]);
      }
    }
    __syncthreads();
  }
#pragma unroll
  for (int mt = 0; mt < 2; ++mt)
#pragma unroll
    for (int nt = 0; nt < 4; ++nt)
#pragma unroll
      for (int r = 0; r < 8; ++r) {
        const int j = wm*32 + mt*16 + half*8 + r;
        const int c = wn*64 + nt*16 + hm;
        dst[(size_t)(rowtile*128 + j) * HQ + ntile*128 + c] = f2bf(acc[mt][nt][r]);
      }
}

// ---- 4) mlm GEMM: x_ln(8192x1024) @ W_mlm^T(1024x32) -> f32 ----------------
__global__ void __launch_bounds__(256)
mlm_gemm(const unsigned short* __restrict__ xln,
         const unsigned short* __restrict__ wm,
         float* __restrict__ out) {
  __shared__ unsigned short As[128][72];
  __shared__ unsigned short Ws[32][72];
  const int t = threadIdx.x, rb = blockIdx.x * 128;
  const int wid = t >> 5, lane = t & 31, half = lane >> 4, hm = lane & 15;
  v8f acc[2];
  const v8f vz = {0.f,0.f,0.f,0.f,0.f,0.f,0.f,0.f};
  acc[0] = vz; acc[1] = vz;

  for (int k0 = 0; k0 < HQ; k0 += 64) {
    for (int pass = 0; pass < 4; ++pass) {
      const int idx = pass * 2048 + t * 8;
      const int j = idx >> 6, col = idx & 63;
      *(us8*)&As[j][col] = *(const us8*)(xln + (size_t)(rb + j) * HQ + k0 + col);
    }
    { const int v = t >> 3, col = (t & 7) * 8;
      *(us8*)&Ws[v][col] = *(const us8*)(wm + (size_t)v * HQ + k0 + col); }
    __syncthreads();
    for (int kk = 0; kk < 64; kk += 32) {
      const int r0 = wid * 16 + hm;
      v16bf a = load_frag(&As[r0][kk + half*8], &As[r0][kk + 16 + half*8]);
#pragma unroll
      for (int nt = 0; nt < 2; ++nt) {
        const unsigned short* bp = &Ws[nt*16 + hm][kk + half*8];
        v16bf bf = load_frag(bp, bp + 16);
        acc[nt] = wmma_bf16(a, bf, acc[nt]);
      }
    }
    __syncthreads();
  }
#pragma unroll
  for (int nt = 0; nt < 2; ++nt)
#pragma unroll
    for (int r = 0; r < 8; ++r)
      out[(size_t)(rb + wid*16 + half*8 + r) * VQ + nt*16 + hm] = acc[nt][r];
}

// ---- 5) Main GEMM: per (b,i): (q_i * k_j) @ W_embed^T -> logits f32 --------
__global__ void __launch_bounds__(256)
logits_gemm(const unsigned short* __restrict__ qb,
            const unsigned short* __restrict__ kb,
            const unsigned short* __restrict__ web,
            float* __restrict__ out) {
  __shared__ unsigned short As[128][72];
  __shared__ unsigned short Ws[128][72];
  const int t = threadIdx.x;
  const int bi = blockIdx.x;           // b*M + i
  const int b = bi >> 7;
  const int vtile = blockIdx.y;
  const int wid = t >> 5, lane = t & 31, half = lane >> 4, hm = lane & 15;
  const int wm = wid & 3, wn = wid >> 2;

  v8f acc[2][4];
  const v8f vz = {0.f,0.f,0.f,0.f,0.f,0.f,0.f,0.f};
  for (int a = 0; a < 2; ++a) for (int c = 0; c < 4; ++c) acc[a][c] = vz;

  for (int k0 = 0; k0 < HQ; k0 += 64) {
    for (int pass = 0; pass < 4; ++pass) {
      const int idx = pass * 2048 + t * 8;
      const int j = idx >> 6, col = idx & 63;
      us8 kv = *(const us8*)(kb + (size_t)(b * MQ + j) * HQ + k0 + col);
      us8 qv = *(const us8*)(qb + (size_t)bi * HQ + k0 + col);
      us8 pr;
#pragma unroll
      for (int e = 0; e < 8; ++e) pr[e] = f2bf(bf2f(kv[e]) * bf2f(qv[e]));
      *(us8*)&As[j][col] = pr;
      *(us8*)&Ws[j][col] = *(const us8*)(web + (size_t)(vtile*128 + j) * HQ + k0 + col);
    }
    __syncthreads();
    for (int kk = 0; kk < 64; kk += 32) {
      const int r0 = wm * 32 + hm, r1 = r0 + 16;
      v16bf a0 = load_frag(&As[r0][kk + half*8], &As[r0][kk + 16 + half*8]);
      v16bf a1 = load_frag(&As[r1][kk + half*8], &As[r1][kk + 16 + half*8]);
#pragma unroll
      for (int nt = 0; nt < 4; ++nt) {
        const unsigned short* bp = &Ws[wn*64 + nt*16 + hm][kk + half*8];
        v16bf bf = load_frag(bp, bp + 16);
        acc[0][nt] = wmma_bf16(a0, bf, acc[0][nt]);
        acc[1][nt] = wmma_bf16(a1, bf, acc[1][nt]);
      }
    }
    __syncthreads();
  }
#pragma unroll
  for (int mt = 0; mt < 2; ++mt)
#pragma unroll
    for (int nt = 0; nt < 4; ++nt)
#pragma unroll
      for (int r = 0; r < 8; ++r) {
        const int j = wm*32 + mt*16 + half*8 + r;
        const int c = wn*64 + nt*16 + hm;
        out[((size_t)bi * MQ + j) * V2 + vtile*128 + c] = acc[mt][nt][r];
      }
}

// ---- 6) masks ---------------------------------------------------------------
__global__ void write_diag(float* __restrict__ out) {
  int i = blockIdx.x * blockDim.x + threadIdx.x;
  if (i < BQ * MQ * MQ) {
    int p = i % (MQ * MQ);
    out[i] = (p / MQ == p % MQ) ? 1.f : 0.f;
  }
}
__global__ void zero_f32(float* __restrict__ p, int n) {
  int i = blockIdx.x * blockDim.x + threadIdx.x;
  if (i < n) p[i] = 0.f;
}
__global__ void scatter_span(const int* __restrict__ midx, float* __restrict__ span) {
  int i = blockIdx.x * blockDim.x + threadIdx.x;
  if (i < BQ * MQ) {
    int b = i / MQ, idx = midx[i];
    if (idx >= 0 && idx < LQ) span[b * LQ + idx] = 1.f;
  }
}
__global__ void write_mask_aa(const float* __restrict__ span, float* __restrict__ out) {
  int i = blockIdx.x * blockDim.x + threadIdx.x;
  if (i < BQ * LQ) out[i] = span[i];
}
__global__ void write_pair(const float* __restrict__ span, float* __restrict__ out) {
  int i = blockIdx.x * blockDim.x + threadIdx.x;
  if (i < BQ * LQ * LQ) {
    int b = i / (LQ * LQ), r = (i / LQ) % LQ, c = i % LQ;
    out[i] = span[b * LQ + r] * span[b * LQ + c];
  }
}

// ---- launcher ---------------------------------------------------------------
extern "C" void kernel_launch(void* const* d_in, const int* in_sizes, int n_in,
                              void* d_out, int out_size, void* d_ws, size_t ws_size,
                              hipStream_t stream) {
  (void)in_sizes; (void)n_in; (void)out_size; (void)ws_size;
  const float* x_enc = (const float*)d_in[0];
  const float* gamma = (const float*)d_in[1];
  const float* beta  = (const float*)d_in[2];
  const float* Wq    = (const float*)d_in[3];
  const float* Wk    = (const float*)d_in[4];
  const float* We    = (const float*)d_in[5];
  const float* Wm    = (const float*)d_in[6];
  const int*   midx  = (const int*)d_in[7];
  float* out = (float*)d_out;

  // workspace layout (bytes); total ~27.4 MB
  char* w = (char*)d_ws;
  unsigned short* xln = (unsigned short*)(w);                    // 8192x1024 bf16
  unsigned short* wqb = (unsigned short*)(w + 16777216);         // 1024x1024 bf16
  unsigned short* wkb = (unsigned short*)(w + 18874368);
  unsigned short* web = (unsigned short*)(w + 20971520);
  unsigned short* wmb = (unsigned short*)(w + 23068672);         // 32x1024 bf16
  unsigned short* qb  = (unsigned short*)(w + 23134208);         // 1024x1024 bf16
  unsigned short* kb  = (unsigned short*)(w + 25231360);
  float*          span = (float*)(w + 27328512);                 // 8192 f32

  // d_out layout (floats, reference return order)
  const size_t off_mlm  = 134217728;   // B*M*M*V2
  const size_t off_diag = off_mlm  + (size_t)BQ * LQ * VQ;   // +262144
  const size_t off_maa  = off_diag + (size_t)BQ * MQ * MQ;   // +131072
  const size_t off_pair = off_maa  + (size_t)BQ * LQ;        // +8192

  ln_kernel<<<BQ * LQ, 256, 0, stream>>>(x_enc, gamma, beta, xln);
  f32_to_bf16<<<512, 256, 0, stream>>>(Wq, wqb, HQ * HQ);
  f32_to_bf16<<<512, 256, 0, stream>>>(Wk, wkb, HQ * HQ);
  f32_to_bf16<<<512, 256, 0, stream>>>(We, web, V2 * HQ);
  f32_to_bf16<<<64, 256, 0, stream>>>(Wm, wmb, VQ * HQ);

  qk_gemm<<<dim3(8, 8, 2), 256, 0, stream>>>(xln, wqb, wkb, midx, qb, kb);
  mlm_gemm<<<64, 256, 0, stream>>>(xln, wmb, out + off_mlm);
  logits_gemm<<<dim3(1024, 8), 256, 0, stream>>>(qb, kb, web, out);

  write_diag<<<(BQ*MQ*MQ + 255) / 256, 256, 0, stream>>>(out + off_diag);
  zero_f32<<<(BQ*LQ + 255) / 256, 256, 0, stream>>>(span, BQ * LQ);
  scatter_span<<<(BQ*MQ + 255) / 256, 256, 0, stream>>>(midx, span);
  write_mask_aa<<<(BQ*LQ + 255) / 256, 256, 0, stream>>>(span, out + off_maa);
  write_pair<<<(BQ*LQ*LQ + 255) / 256, 256, 0, stream>>>(span, out + off_pair);
}
